// MultiHeadSelfAttention_88983132439182
// MI455X (gfx1250) — matrix operand order repaired and checked by reading, not compiled, not run on hardware
//
#include <hip/hip_runtime.h>
#include <hip/hip_bf16.h>

typedef __bf16 bf16;
typedef __attribute__((ext_vector_type(16))) __bf16 v16bf;
typedef __attribute__((ext_vector_type(8)))  __bf16 bf16x8;
typedef __attribute__((ext_vector_type(4)))  __bf16 bf16x4;
typedef __attribute__((ext_vector_type(8)))  float   v8f;
typedef __attribute__((ext_vector_type(4)))  int     v4i;

#define D_MODEL 1024
#define N_HEADS 16
#define D_K     64
#define SEQ     2048
#define BATCH   2
#define M_TOTAL (BATCH*SEQ)   // 4096
#define STILES  (SEQ/64)      // 32 s-tiles in attention

// ---------------------------------------------------------------------------
// CDNA5 async global->LDS DMA (ASYNCcnt path). Builtins probe-confirmed to
// exist here; parameter type is generic v4i* per compiler diagnostic.
// ---------------------------------------------------------------------------
#if defined(__has_builtin)
#  if __has_builtin(__builtin_amdgcn_global_load_async_to_lds_b128) && \
      __has_builtin(__builtin_amdgcn_s_wait_asynccnt)
#    define USE_ASYNC_LDS 1
#  endif
#endif
#ifndef USE_ASYNC_LDS
#  define USE_ASYNC_LDS 0
#endif

#if USE_ASYNC_LDS
#  define WAIT_ASYNC(n) __builtin_amdgcn_s_wait_asynccnt(n)
#else
#  define WAIT_ASYNC(n) ((void)0)
#endif

// ---------------------------------------------------------------------------
// WMMA fragment helpers (layouts per CDNA5 ISA §7.12.2, wave32)
//  A (16x32 bf16):  lane lo = row M, hi=lane>>4 selects K-halves:
//    elements 0..7  = K[hi*8 .. hi*8+7],  elements 8..15 = K[hi*8+16 .. +23]
//  B (32x16 bf16):  lane lo = column N, same K order as A:
//    elements 0..7  = K[hi*8 .. hi*8+7],  elements 8..15 = K[hi*8+16 .. +23]
//    (LDS tiles for B are stored [N][K]-major -> two 16B reads)
// ---------------------------------------------------------------------------
__device__ __forceinline__ v16bf load_frag_a(const bf16* p) {
  v16bf a;
  *((bf16x8*)&a)       = *(const bf16x8*)(p);
  *(((bf16x8*)&a) + 1) = *(const bf16x8*)(p + 16);
  return a;
}
__device__ __forceinline__ v16bf load_frag_b(const bf16* p) {
  v16bf b;
  *((bf16x8*)&b)       = *(const bf16x8*)(p);
  *(((bf16x8*)&b) + 1) = *(const bf16x8*)(p + 16);
  return b;
}
__device__ __forceinline__ v8f wmma_bf16(v16bf a, v16bf b, v8f c) {
  return __builtin_amdgcn_wmma_f32_16x16x32_bf16(false, a, false, b,
                                                 (short)0, c, false, false);
}

// ---------------------------------------------------------------------------
// GEMM: C[M,N] = A[M,K] (fp32 or bf16) @ W[K,N] (fp32->bf16) + bias
// Block tile 128x128, 256 threads (8 waves), wave tile 64x32 (4x2 WMMA tiles)
// ---------------------------------------------------------------------------
#define LDA_S 40   // bf16 elems/row in LDS A tile (32 + pad, 80B = 16B-aligned)
#define LDB_S 40   // bf16 elems/row in LDS Bt tile

template<bool A_BF16, bool SPLIT_HEADS, bool OUT_F32>
__global__ __launch_bounds__(256)
void gemm_kernel(const void* __restrict__ Ap, const float* __restrict__ W,
                 const float* __restrict__ bias, void* __restrict__ Outp)
{
  __shared__ __align__(16) bf16 As[128 * LDA_S];
  __shared__ __align__(16) bf16 Bs[128 * LDB_S];   // stored [n][k]

  const int K = D_MODEL, N = D_MODEL;
  const int tid = threadIdx.x, lane = tid & 31, wave = tid >> 5;
  const int wm = wave >> 2, wn = wave & 3;          // 2 x 4 wave grid
  const int bn = blockIdx.x * 128, bm = blockIdx.y * 128;
  const int lo = lane & 15, hi = lane >> 4;

  const v8f vzero = {};
  v8f acc[4][2];
  #pragma unroll
  for (int mt = 0; mt < 4; ++mt)
    #pragma unroll
    for (int nt = 0; nt < 2; ++nt) acc[mt][nt] = vzero;

  for (int kk = 0; kk < K; kk += 32) {
    // L2-warm the next K-step tiles while this one is staged/computed
    if (kk + 32 < K) {
      if constexpr (A_BF16)
        __builtin_prefetch((const bf16*)Ap + (size_t)(bm + (tid >> 1)) * K + kk + 32 + (tid & 1) * 32, 0, 1);
      else
        __builtin_prefetch((const float*)Ap + (size_t)(bm + (tid >> 1)) * K + kk + 32 + (tid & 1) * 16, 0, 1);
      __builtin_prefetch(W + (size_t)(kk + 32 + (tid >> 3)) * N + bn + (tid & 7) * 16, 0, 1);
    }
    // ---- stage A tile: 128 x 32, convert to bf16 ----
    #pragma unroll
    for (int it = 0; it < 4; ++it) {
      int task = tid + it * 256;           // 1024 tasks of 4 elems
      int row = task >> 3, c4 = (task & 7) * 4;
      if constexpr (A_BF16) {
        const bf16* ag = (const bf16*)Ap + (size_t)(bm + row) * K + kk + c4;
        *(bf16x4*)&As[row * LDA_S + c4] = *(const bf16x4*)ag;
      } else {
        float4 v = *(const float4*)((const float*)Ap + (size_t)(bm + row) * K + kk + c4);
        bf16* dst = &As[row * LDA_S + c4];
        dst[0] = (bf16)v.x; dst[1] = (bf16)v.y;
        dst[2] = (bf16)v.z; dst[3] = (bf16)v.w;
      }
    }
    // ---- stage W tile transposed: Bs[n][k], 32 x 128 source ----
    #pragma unroll
    for (int it = 0; it < 4; ++it) {
      int task = tid + it * 256;
      int r = task >> 5, c4 = (task & 31) * 4;
      float4 v = *(const float4*)(W + (size_t)(kk + r) * N + bn + c4);
      Bs[(c4 + 0) * LDB_S + r] = (bf16)v.x;
      Bs[(c4 + 1) * LDB_S + r] = (bf16)v.y;
      Bs[(c4 + 2) * LDB_S + r] = (bf16)v.z;
      Bs[(c4 + 3) * LDB_S + r] = (bf16)v.w;
    }
    __syncthreads();

    v16bf af[4], bfr[2];
    #pragma unroll
    for (int mt = 0; mt < 4; ++mt)
      af[mt] = load_frag_a(&As[(wm * 64 + mt * 16 + lo) * LDA_S + hi * 8]);
    #pragma unroll
    for (int nt = 0; nt < 2; ++nt)
      bfr[nt] = load_frag_b(&Bs[(wn * 32 + nt * 16 + lo) * LDB_S + hi * 8]);
    #pragma unroll
    for (int mt = 0; mt < 4; ++mt)
      #pragma unroll
      for (int nt = 0; nt < 2; ++nt)
        acc[mt][nt] = wmma_bf16(af[mt], bfr[nt], acc[mt][nt]);
    __syncthreads();
  }

  // ---- epilogue: bias + store ----
  #pragma unroll
  for (int nt = 0; nt < 2; ++nt) {
    int gcol = bn + wn * 32 + nt * 16 + lo;
    float bv = bias[gcol];
    #pragma unroll
    for (int mt = 0; mt < 4; ++mt) {
      #pragma unroll
      for (int v = 0; v < 8; ++v) {
        int grow = bm + wm * 64 + mt * 16 + hi * 8 + v;   // C layout: M = v + hi*8
        float val = acc[mt][nt][v] + bv;
        if constexpr (SPLIT_HEADS) {
          int h = gcol >> 6, d = gcol & 63;
          int b = grow >> 11, t = grow & 2047;
          ((bf16*)Outp)[(((size_t)(b * N_HEADS + h) * SEQ) + t) * D_K + d] = (bf16)val;
        } else if constexpr (OUT_F32) {
          ((float*)Outp)[(size_t)grow * N + gcol] = val;
        } else {
          ((bf16*)Outp)[(size_t)grow * N + gcol] = (bf16)val;
        }
      }
    }
  }
}

// ---------------------------------------------------------------------------
// Stage one 64x64 bf16 tile (global, row stride D_K) into LDS (row stride
// AT_LDB). Async DMA when available (4 b128 per thread = ASYNCcnt +4/wave).
// ---------------------------------------------------------------------------
#define AT_LDB  72   // bf16 row stride (64 + pad, 144B = 16B-aligned)
#define AT_LDSF 68   // f32  row stride for score tile

__device__ __forceinline__ void stage_tile(const bf16* __restrict__ g,
                                           bf16* l, int tid) {
  #pragma unroll
  for (int it = 0; it < 4; ++it) {
    int task = tid + it * 128;
    int row = task >> 3, c8 = (task & 7) * 8;
#if USE_ASYNC_LDS
    __builtin_amdgcn_global_load_async_to_lds_b128(
        (v4i*)const_cast<bf16*>(g + row * D_K + c8),
        (v4i*)(l + row * AT_LDB + c8),
        0, 0);
#else
    *(bf16x8*)&l[row * AT_LDB + c8] = *(const bf16x8*)&g[row * D_K + c8];
#endif
  }
}

// ---------------------------------------------------------------------------
// Flash attention: one WG per (q-tile of 64 rows, b*h). 128 threads = 4 waves,
// each wave owns 16 q rows. K/V stream in 64-column tiles, double-buffered:
// K via async DMA, V via registers (transposed into LDS under softmax cover).
// ---------------------------------------------------------------------------
__global__ __launch_bounds__(128)
void attn_kernel(const bf16* __restrict__ Q, const bf16* __restrict__ K,
                 const bf16* __restrict__ V, bf16* __restrict__ Ctx)
{
  __shared__ __align__(16) bf16  QP[64 * AT_LDB];      // Q tile, then P (aliased)
  __shared__ __align__(16) bf16  Ks[2][64 * AT_LDB];   // [s][d]  (B-frag, Q.K^T)
  __shared__ __align__(16) bf16  Vt[2][64 * AT_LDB];   // [d][s]  (B-frag, P.V)
  __shared__ __align__(16) float Sf[64 * AT_LDSF];     // scores, f32
  __shared__ float rowM[64], rowL[64], rowA[64];
  // LDS: 9216 + 18432 + 18432 + 17408 + 768 = 64256 B (< 64 KB)

  const int tid = threadIdx.x, lane = tid & 31, wave = tid >> 5;
  const int lo = lane & 15, hi = lane >> 4;
  const int qt = blockIdx.x, bh = blockIdx.y;
  const v8f vzero = {};

  const bf16* Qg    = Q + ((size_t)bh * SEQ + qt * 64) * D_K;
  const bf16* Kbase = K + (size_t)bh * SEQ * D_K;
  const bf16* Vbase = V + (size_t)bh * SEQ * D_K;

  // ---- prologue: Q + K0 async in flight, V0 via registers ----
  stage_tile(Qg, QP, tid);           // 4 asyncs (Q)
  stage_tile(Kbase, Ks[0], tid);     // 4 asyncs (K tile 0)
  {
    #pragma unroll
    for (int it = 0; it < 4; ++it) {
      int task = tid + it * 128, row = task >> 3, c8 = (task & 7) * 8;
      bf16x8 vv = *(const bf16x8*)&Vbase[row * D_K + c8];
      #pragma unroll
      for (int j = 0; j < 8; ++j) Vt[0][(c8 + j) * AT_LDB + row] = vv[j];
    }
  }
  if (tid < 64) { rowM[tid] = -1e30f; rowL[tid] = 0.f; }
  WAIT_ASYNC(4);        // drain Q (in-order; K0 may stay in flight)
  __syncthreads();      // Q, Vt[0], stats visible to all waves

  v16bf qf[2];
  #pragma unroll
  for (int ks = 0; ks < 2; ++ks)
    qf[ks] = load_frag_a(&QP[(wave * 16 + lo) * AT_LDB + ks * 32 + hi * 8]);

  v8f o[4];
  #pragma unroll
  for (int nt = 0; nt < 4; ++nt) o[nt] = vzero;

  for (int i = 0; i < STILES; ++i) {
    const int cur = i & 1, nxt = cur ^ 1;
    const bool more = (i + 1) < STILES;

    // issue next K tile DMA into the other buffer (its last readers finished
    // before the previous end-of-iteration barrier), then drain tile i only
    if (more) {
      stage_tile(Kbase + (size_t)(i + 1) * 64 * D_K, Ks[nxt], tid);
      WAIT_ASYNC(4);    // K(i) done for this wave; K(i+1) stays in flight
    } else {
      WAIT_ASYNC(0);
    }
    if (i + 2 < STILES) {
      __builtin_prefetch(Kbase + (size_t)(i + 2) * 64 * D_K + tid * 64, 0, 1);
      __builtin_prefetch(Vbase + (size_t)(i + 2) * 64 * D_K + tid * 64, 0, 1);
    }
    __syncthreads();    // all waves' K(i) DMA + last iter's Vt writes visible

    // next V tile -> registers (latency hidden under S + softmax)
    bf16x8 vn[4];
    if (more) {
      #pragma unroll
      for (int it = 0; it < 4; ++it) {
        int task = tid + it * 128, row = task >> 3, c8 = (task & 7) * 8;
        vn[it] = *(const bf16x8*)&Vbase[((size_t)(i + 1) * 64 + row) * D_K + c8];
      }
    }

    // ---- S = Q K^T (wave: 16 x 64) ----
    v8f sa[4];
    #pragma unroll
    for (int nt = 0; nt < 4; ++nt) {
      sa[nt] = vzero;
      #pragma unroll
      for (int ks = 0; ks < 2; ++ks) {
        v16bf b = load_frag_b(&Ks[cur][(nt * 16 + lo) * AT_LDB + ks * 32 + hi * 8]);
        sa[nt] = wmma_bf16(qf[ks], b, sa[nt]);
      }
    }
    #pragma unroll
    for (int nt = 0; nt < 4; ++nt)
      #pragma unroll
      for (int v = 0; v < 8; ++v)
        Sf[(wave * 16 + hi * 8 + v) * AT_LDSF + nt * 16 + lo] = sa[nt][v] * 0.125f;
    __syncthreads();

    // ---- online softmax (one thread per q row), P -> QP (bf16) ----
    if (tid < 64) {
      float mold = rowM[tid], lold = rowL[tid];
      const float* srow = &Sf[tid * AT_LDSF];
      float mt = -1e30f;
      #pragma unroll 8
      for (int j = 0; j < 64; ++j) mt = fmaxf(mt, srow[j]);
      float mnew = fmaxf(mold, mt);
      float alpha = __expf(mold - mnew);
      float sum = 0.f;
      bf16* prow = &QP[tid * AT_LDB];
      #pragma unroll 8
      for (int j = 0; j < 64; ++j) {
        float p = __expf(srow[j] - mnew);
        sum += p;
        prow[j] = (bf16)p;
      }
      rowM[tid] = mnew;
      rowL[tid] = lold * alpha + sum;
      rowA[tid] = alpha;
    }
    __syncthreads();

    // scatter next V tile into the other Vt buffer (readers come after the
    // next iteration's first barrier; our ds writes retire at the end barrier)
    if (more) {
      #pragma unroll
      for (int it = 0; it < 4; ++it) {
        int task = tid + it * 128, row = task >> 3, c8 = (task & 7) * 8;
        #pragma unroll
        for (int j = 0; j < 8; ++j) Vt[nxt][(c8 + j) * AT_LDB + row] = vn[it][j];
      }
    }

    // ---- rescale O, then O += P V ----
    float al[8];
    #pragma unroll
    for (int v = 0; v < 8; ++v) al[v] = rowA[wave * 16 + hi * 8 + v];
    #pragma unroll
    for (int nt = 0; nt < 4; ++nt)
      #pragma unroll
      for (int v = 0; v < 8; ++v) o[nt][v] *= al[v];

    v16bf pf[2];
    #pragma unroll
    for (int ks = 0; ks < 2; ++ks)
      pf[ks] = load_frag_a(&QP[(wave * 16 + lo) * AT_LDB + ks * 32 + hi * 8]);
    #pragma unroll
    for (int nt = 0; nt < 4; ++nt)
      #pragma unroll
      for (int ks = 0; ks < 2; ++ks) {
        v16bf b = load_frag_b(&Vt[cur][(nt * 16 + lo) * AT_LDB + ks * 32 + hi * 8]);
        o[nt] = wmma_bf16(pf[ks], b, o[nt]);
      }
    __syncthreads();   // protect Ks/Vt/QP before next tile's writes
  }

  // ---- normalize and store context as [B*T, D] bf16 ----
  float linv[8];
  #pragma unroll
  for (int v = 0; v < 8; ++v) linv[v] = 1.f / rowL[wave * 16 + hi * 8 + v];
  const int b = bh >> 4, h = bh & 15;
  #pragma unroll
  for (int nt = 0; nt < 4; ++nt)
    #pragma unroll
    for (int v = 0; v < 8; ++v) {
      int t = qt * 64 + wave * 16 + hi * 8 + v;
      int d = nt * 16 + lo;
      Ctx[((size_t)(b * SEQ + t)) * D_MODEL + h * D_K + d] = (bf16)(o[nt][v] * linv[v]);
    }
}

// ---------------------------------------------------------------------------
extern "C" void kernel_launch(void* const* d_in, const int* in_sizes, int n_in,
                              void* d_out, int out_size, void* d_ws, size_t ws_size,
                              hipStream_t stream) {
  (void)in_sizes; (void)n_in; (void)out_size; (void)ws_size;
  const float* x  = (const float*)d_in[0];
  const float* Wq = (const float*)d_in[1];
  const float* bq = (const float*)d_in[2];
  const float* Wk = (const float*)d_in[3];
  const float* bk = (const float*)d_in[4];
  const float* Wv = (const float*)d_in[5];
  const float* bv = (const float*)d_in[6];
  const float* Wo = (const float*)d_in[7];
  const float* bo = (const float*)d_in[8];

  const size_t elems = (size_t)BATCH * N_HEADS * SEQ * D_K;  // 4 Mi elements
  bf16* Qb = (bf16*)d_ws;          // [B*H, T, d_k] bf16
  bf16* Kb = Qb + elems;
  bf16* Vb = Kb + elems;
  bf16* Cb = Vb + elems;           // context [B*T, D] bf16
  // workspace use: 4 * 4Mi * 2B = 33.6 MB

  dim3 gg(D_MODEL / 128, M_TOTAL / 128);   // (8, 32)
  gemm_kernel<false, true,  false><<<gg, 256, 0, stream>>>(x,  Wq, bq, Qb);
  gemm_kernel<false, true,  false><<<gg, 256, 0, stream>>>(x,  Wk, bk, Kb);
  gemm_kernel<false, true,  false><<<gg, 256, 0, stream>>>(x,  Wv, bv, Vb);

  attn_kernel<<<dim3(SEQ / 64, BATCH * N_HEADS), 128, 0, stream>>>(Qb, Kb, Vb, Cb);

  gemm_kernel<true,  false, true ><<<gg, 256, 0, stream>>>(Cb, Wo, bo, d_out);
}
